// MultiHeadSelfAttention_25400436588667
// MI455X (gfx1250) — matrix operand order repaired and checked by reading, not compiled, not run on hardware
//
#include <hip/hip_runtime.h>
#include <hip/hip_bf16.h>
#include <math.h>

// -------- CDNA5 WMMA / vector types --------
typedef __attribute__((ext_vector_type(16))) __bf16 v16bf;
typedef __attribute__((ext_vector_type(8)))  __bf16 v8bf;
typedef __attribute__((ext_vector_type(8)))  float  v8f;
typedef __attribute__((ext_vector_type(4)))  float  v4f;

// -------- problem constants --------
constexpr int BATCH = 2;
constexpr int L     = 2048;
constexpr int D     = 1024;
constexpr int H     = 16;
constexpr int DH    = 64;
constexpr int MTOK  = BATCH * L;                          // 4096 token rows
constexpr size_t OUT_ELEMS  = (size_t)MTOK * D;           // 4,194,304
constexpr size_t QKV_ELEMS  = (size_t)BATCH * H * L * DH; // 4,194,304 (bf16 each)

// LDS score-block row stride (floats). 2052 % 64 == 4 -> consecutive rows
// shift LDS banks by 4, rows 8 apart shift by 32: the two EXEC-halves of a
// wave hit disjoint bank groups on stores, and per-lane-row reads in the PV
// phase are conflict-free. 2052 % 4 == 0 keeps float4 (b128) alignment.
constexpr int SROW = 2052;
constexpr int SMEM_BYTES = 16 * SROW * 4; // 131,328 B -> 2 WGs / 320KB WGP

__device__ __forceinline__ void cvt4(v16bf& v, int base, float4 f) {
    v[base + 0] = (__bf16)f.x; v[base + 1] = (__bf16)f.y;
    v[base + 2] = (__bf16)f.z; v[base + 3] = (__bf16)f.w;
}

// A-operand (16x32 MxK, bf16) built from a row-major fp32 matrix.
// ISA layout: lane m = lane&15; element e -> K = (e>>3)*16 + half*8 + (e&7)
__device__ __forceinline__ v16bf load_a_f32(const float* __restrict__ base,
                                            int ldk, int m, int k0, int half) {
    const float* r = base + (size_t)m * ldk + k0 + half * 8;
    float4 f0 = *(const float4*)(r + 0);
    float4 f1 = *(const float4*)(r + 4);
    float4 f2 = *(const float4*)(r + 16);
    float4 f3 = *(const float4*)(r + 20);
    v16bf a;
    cvt4(a, 0, f0); cvt4(a, 4, f1); cvt4(a, 8, f2); cvt4(a, 12, f3);
    return a;
}

// B-operand (32x16 KxN, bf16) from row-major fp32 W[n][k] (contiguous in k).
// ISA layout: lane n = lane&15; element e -> K = (e>>3)*16 + half*8 + (e&7)
__device__ __forceinline__ v16bf load_b_f32(const float* __restrict__ base,
                                            int ldk, int n, int k0, int half) {
    const float* r = base + (size_t)n * ldk + k0 + half * 8;
    float4 g0 = *(const float4*)(r + 0);
    float4 g1 = *(const float4*)(r + 4);
    float4 g2 = *(const float4*)(r + 16);
    float4 g3 = *(const float4*)(r + 20);
    v16bf b;
    cvt4(b, 0, g0); cvt4(b, 4, g1); cvt4(b, 8, g2); cvt4(b, 12, g3);
    return b;
}

__device__ __forceinline__ v16bf load_a_bf16(const __bf16* __restrict__ base,
                                             int ldk, int m, int k0, int half) {
    const __bf16* r = base + (size_t)m * ldk + k0 + half * 8;
    v8bf lo = *(const v8bf*)(r);
    v8bf hi = *(const v8bf*)(r + 16);
    v16bf a;
#pragma unroll
    for (int i = 0; i < 8; ++i) { a[i] = lo[i]; a[8 + i] = hi[i]; }
    return a;
}

__device__ __forceinline__ v16bf load_b_bf16(const __bf16* __restrict__ base,
                                             int ldk, int n, int k0, int half) {
    const __bf16* r = base + (size_t)n * ldk + k0 + half * 8;
    v8bf lo = *(const v8bf*)(r);
    v8bf hi = *(const v8bf*)(r + 16);
    v16bf b;
#pragma unroll
    for (int i = 0; i < 8; ++i) { b[i] = lo[i]; b[8 + i] = hi[i]; }
    return b;
}

// ---------------------------------------------------------------------------
// Generic Y = X (fp32 [4096,1024]) @ W^T (fp32 [1024,1024]) + bias
// mode 0: bf16 out, head-split [B,H,L,DH]                (Q, K)
// mode 2: bf16 out, head-split transposed [B,H,DH,L]     (V)
// mode 1: fp32 out, row-major [4096,1024]                (final projection)
// ---------------------------------------------------------------------------
__global__ __launch_bounds__(256) void gemm_xwT_kernel(
    const float* __restrict__ X, const float* __restrict__ W,
    const float* __restrict__ bias, float* __restrict__ outF,
    __bf16* __restrict__ outH, int mode) {
    const int gid  = blockIdx.x * 8 + (threadIdx.x >> 5);
    const int lane = threadIdx.x & 31;
    const int half = lane >> 4;
    const int lm   = lane & 15;
    const int n0   = (gid & 63) * 16;   // N = 1024 -> 64 tiles
    const int m0   = (gid >> 6) * 16;   // M = 4096 -> 256 tiles

    v8f acc = {};
#pragma unroll 4
    for (int k0 = 0; k0 < D; k0 += 32) {
        v16bf a = load_a_f32(X, D, m0 + lm, k0, half);
        v16bf b = load_b_f32(W, D, n0 + lm, k0, half);
        acc = __builtin_amdgcn_wmma_f32_16x16x32_bf16(
            false, a, false, b, (short)0, acc, false, false);
    }

    const int n  = n0 + lm;
    const float bb = bias[n];
    if (mode == 1) {
#pragma unroll
        for (int r = 0; r < 8; ++r) {
            int m = m0 + r + half * 8;
            outF[(size_t)m * D + n] = acc[r] + bb;
        }
    } else if (mode == 0) {
        const int h = n >> 6, d = n & 63;
#pragma unroll
        for (int r = 0; r < 8; ++r) {
            int m = m0 + r + half * 8;
            int b_ = m >> 11, l = m & (L - 1);
            outH[(((size_t)(b_ * H + h) * L + l) << 6) + d] = (__bf16)(acc[r] + bb);
        }
    } else { // mode 2: transposed per-head V: [B,H,DH,L]
        const int h = n >> 6, d = n & 63;
#pragma unroll
        for (int r = 0; r < 8; ++r) {
            int m = m0 + r + half * 8;
            int b_ = m >> 11, l = m & (L - 1);
            outH[(((size_t)(b_ * H + h) << 6) + d) * L + l] = (__bf16)(acc[r] + bb);
        }
    }
}

// ---------------------------------------------------------------------------
// Fused attention: one workgroup (8 waves) per (bh, 16-query block).
//  Phase 1: S = 0.125 * Q Kt, WMMA, each wave owns a 16x256 strip -> LDS.
//  Phase 2: row softmax in LDS (16 lanes/row, 4 consecutive cols per lane,
//           shfl-xor reductions); normalized weights stream to d_out exactly
//           once via b128 nontemporal stores (full-line writes, no L2
//           pollution -- P is never re-read from global).
//  Phase 3: ctx = P V, waves 0..3 (one 16x16 Dh tile each), A-operand read
//           from LDS (fp32->bf16), B-operand from transposed bf16 V.
// Attn-region HBM traffic: single 537 MB write (vs 2.15 GB unfused).
// ---------------------------------------------------------------------------
__global__ __launch_bounds__(256) void attn_fused_kernel(
    const __bf16* __restrict__ Q, const __bf16* __restrict__ K,
    const __bf16* __restrict__ Vt, float* __restrict__ attn,
    float* __restrict__ ctx) {
    extern __shared__ float S[]; // [16][SROW]

    const int tid  = threadIdx.x;
    const int wave = tid >> 5;
    const int lane = tid & 31;
    const int half = lane >> 4;
    const int lm   = lane & 15;

    const int qt = blockIdx.x & 127;   // 128 q-blocks
    const int bh = blockIdx.x >> 7;    // B*H = 32
    const int q0 = qt * 16;

    const __bf16* Qb = Q  + (size_t)bh * L * DH;
    const __bf16* Kb = K  + (size_t)bh * L * DH;
    const __bf16* Vb = Vt + (size_t)bh * DH * L;

    // ---- Phase 1: scores into LDS ----
    v16bf a0 = load_a_bf16(Qb, DH, q0 + lm, 0, half);
    v16bf a1 = load_a_bf16(Qb, DH, q0 + lm, 32, half);
#pragma unroll 4
    for (int t = 0; t < 16; ++t) {
        const int n0 = wave * 256 + t * 16;
        v16bf b0 = load_b_bf16(Kb, DH, n0 + lm, 0, half);
        v16bf b1 = load_b_bf16(Kb, DH, n0 + lm, 32, half);
        v8f acc = {};
        acc = __builtin_amdgcn_wmma_f32_16x16x32_bf16(false, a0, false, b0,
                                                      (short)0, acc, false, false);
        acc = __builtin_amdgcn_wmma_f32_16x16x32_bf16(false, a1, false, b1,
                                                      (short)0, acc, false, false);
#pragma unroll
        for (int r = 0; r < 8; ++r)
            S[(r + half * 8) * SROW + n0 + lm] = acc[r] * 0.125f; // 1/sqrt(64)
    }
    __syncthreads();

    // ---- Phase 2: softmax (16 lanes per row, 4-wide vector per lane) ----
    {
        const int g  = tid >> 4;   // row 0..15
        const int tl = tid & 15;   // lane-in-row; owns cols tl*4+j*64 .. +3
        float* row = S + g * SROW;

        float mx = -3.4e38f;
        for (int j = 0; j < L / 64; ++j) {
            v4f f = *(const v4f*)(row + j * 64 + tl * 4);
            mx = fmaxf(mx, fmaxf(fmaxf(f.x, f.y), fmaxf(f.z, f.w)));
        }
#pragma unroll
        for (int s = 8; s > 0; s >>= 1) mx = fmaxf(mx, __shfl_xor(mx, s, 16));

        float sum = 0.f;
        for (int j = 0; j < L / 64; ++j) {
            v4f f = *(const v4f*)(row + j * 64 + tl * 4);
            f.x = __expf(f.x - mx); f.y = __expf(f.y - mx);
            f.z = __expf(f.z - mx); f.w = __expf(f.w - mx);
            *(v4f*)(row + j * 64 + tl * 4) = f;
            sum += (f.x + f.y) + (f.z + f.w);
        }
#pragma unroll
        for (int s = 8; s > 0; s >>= 1) sum += __shfl_xor(sum, s, 16);
        const float inv = 1.0f / sum;

        float* grow = attn + ((size_t)bh * L + q0 + g) * L;
        for (int j = 0; j < L / 64; ++j) {
            v4f f = *(const v4f*)(row + j * 64 + tl * 4);
            f *= inv;
            *(v4f*)(row + j * 64 + tl * 4) = f;
            __builtin_nontemporal_store(f, (v4f*)(grow + j * 64 + tl * 4));
        }
    }
    __syncthreads();

    // ---- Phase 3: ctx = P @ V (waves 0..3, one Dh 16-col tile each) ----
    if (wave < 4) {
        v8f acc = {};
        for (int k0 = 0; k0 < L; k0 += 32) {
            const float* r0 = S + lm * SROW + k0 + half * 8;
            float4 f0 = *(const float4*)(r0 + 0);
            float4 f1 = *(const float4*)(r0 + 4);
            float4 f2 = *(const float4*)(r0 + 16);
            float4 f3 = *(const float4*)(r0 + 20);
            v16bf a;
            cvt4(a, 0, f0); cvt4(a, 4, f1); cvt4(a, 8, f2); cvt4(a, 12, f3);
            v16bf b = load_b_bf16(Vb, L, wave * 16 + lm, k0, half);
            acc = __builtin_amdgcn_wmma_f32_16x16x32_bf16(
                false, a, false, b, (short)0, acc, false, false);
        }
        const int b_ = bh / H, h = bh % H;
#pragma unroll
        for (int r = 0; r < 8; ++r) {
            int q = q0 + r + half * 8;
            ctx[((size_t)b_ * L + q) * D + h * 64 + wave * 16 + lm] = acc[r];
        }
    }
}

// ---------------------------------------------------------------------------
// Inputs (setup_inputs order): x, Wq, bq, Wk, bk, Wv, bv, Wo, bo  (all fp32)
// d_out: [out (4096*1024 f32)] ++ [attn_weights (2*16*2048*2048 f32)]
// d_ws : >= 40 MB scratch (Q,K bf16 head-split; V bf16 transposed; ctx fp32)
// ---------------------------------------------------------------------------
extern "C" void kernel_launch(void* const* d_in, const int* in_sizes, int n_in,
                              void* d_out, int out_size, void* d_ws, size_t ws_size,
                              hipStream_t stream) {
    (void)in_sizes; (void)n_in; (void)out_size; (void)ws_size;

    const float* x  = (const float*)d_in[0];
    const float* Wq = (const float*)d_in[1];
    const float* bq = (const float*)d_in[2];
    const float* Wk = (const float*)d_in[3];
    const float* bk = (const float*)d_in[4];
    const float* Wv = (const float*)d_in[5];
    const float* bv = (const float*)d_in[6];
    const float* Wo = (const float*)d_in[7];
    const float* bo = (const float*)d_in[8];

    float* outF = (float*)d_out;
    float* attn = outF + OUT_ELEMS;

    __bf16* Qp  = (__bf16*)d_ws;
    __bf16* Kp  = Qp + QKV_ELEMS;
    __bf16* Vt  = Kp + QKV_ELEMS;
    float*  ctx = (float*)(Vt + QKV_ELEMS);

    // QKV projections: (4096/16)*(1024/16)=16384 wave-tiles / 8 = 2048 blocks
    gemm_xwT_kernel<<<2048, 256, 0, stream>>>(x, Wq, bq, nullptr, Qp, 0);
    gemm_xwT_kernel<<<2048, 256, 0, stream>>>(x, Wk, bk, nullptr, Kp, 0);
    gemm_xwT_kernel<<<2048, 256, 0, stream>>>(x, Wv, bv, nullptr, Vt, 2);

    // Fused scores+softmax+PV: one WG per (bh, q-block) = 32*128 = 4096 WGs
    attn_fused_kernel<<<4096, 256, SMEM_BYTES, stream>>>(Qp, Kp, Vt, attn, ctx);

    // Final projection -> fp32 out
    gemm_xwT_kernel<<<2048, 256, 0, stream>>>(ctx, Wo, bo, outF, nullptr, 1);
}
